// GLTEncoder_24369644438005
// MI455X (gfx1250) — matrix operand order repaired and checked by reading, not compiled, not run on hardware
//
#include <hip/hip_runtime.h>
#include <hip/hip_bf16.h>

// ---------------------------------------------------------------------------
// GLT encoder for MI455X (gfx1250, wave32, WMMA).
// Heavy GEMMs run as bf16 WMMA (v_wmma_f32_16x16x32_bf16) with f32 accum.
// Fragments are loaded straight from global memory (K-contiguous rows), no
// LDS staging needed since both A (rows) and W (N,K row-major == B^T) have
// contiguous K.
// ---------------------------------------------------------------------------

typedef __bf16 bf16;
typedef __bf16 bf16x16 __attribute__((ext_vector_type(16)));
typedef float f32x8 __attribute__((ext_vector_type(8)));

#define DD 768
#define DD2 (768 * 768)

__device__ __forceinline__ float geluf(float x) {
    return 0.5f * x * (1.0f + erff(x * 0.70710678118654752f));
}
__device__ __forceinline__ float sigmoidf_(float x) {
    return 1.0f / (1.0f + __expf(-x));
}

// A-fragment (16x32 bf16, MxK): lane l (l16 = l&15, hi = l>>4) holds row
// M = l16; K chunks {hi*8 .. hi*8+7} and {16+hi*8 .. 16+hi*8+7}.
__device__ __forceinline__ bf16x16 load_a_frag(const bf16* base, int ld,
                                               int row, int kb, int hi) {
    const bf16* p = base + (size_t)row * ld + kb + hi * 8;
    union { uint4 u[2]; bf16x16 v; } f;
    f.u[0] = *(const uint4*)p;
    f.u[1] = *(const uint4*)(p + 16);
    return f.v;
}

// B-fragment (32x16 bf16, KxN) from W stored (N,K) row-major: lane group hi
// holds K chunks {hi*8 .. hi*8+7} and {16+hi*8 .. 16+hi*8+7} (matches A),
// column N = l16.  Two uint4 loads.
__device__ __forceinline__ bf16x16 load_b_frag(const bf16* w, int ld,
                                               int col, int kb, int hi) {
    const bf16* p = w + (size_t)col * ld + kb + hi * 8;
    union { uint4 u[2]; bf16x16 v; } f;
    f.u[0] = *(const uint4*)p;
    f.u[1] = *(const uint4*)(p + 16);
    return f.v;
}

// ---------------------------------------------------------------------------
// Elementwise converts
// ---------------------------------------------------------------------------
__global__ void k_f32_to_bf16(const float* __restrict__ src,
                              bf16* __restrict__ dst, long n) {
    long i = (long)blockIdx.x * blockDim.x + threadIdx.x;
    if (i < n) dst[i] = (bf16)src[i];
}

// out[j*768+n] = in[n*768+j]  (transpose-convert, for ff3^T)
__global__ void k_transpose_bf16(const float* __restrict__ src,
                                 bf16* __restrict__ dst) {
    long i = (long)blockIdx.x * blockDim.x + threadIdx.x;
    if (i < (long)DD2) {
        int j = (int)(i / DD), n = (int)(i % DD);
        dst[i] = (bf16)src[(size_t)n * DD + j];
    }
}

// ---------------------------------------------------------------------------
// Stage 1: per pair-row gating weights + bf16 pack of hs_l / hs_r
// one block per row m in [0, 32768)
// ---------------------------------------------------------------------------
__device__ __forceinline__ float block_reduce256(float v, float* red) {
    int t = threadIdx.x;
    red[t] = v;
    __syncthreads();
    for (int s = 128; s > 0; s >>= 1) {
        if (t < s) red[t] += red[t + s];
        __syncthreads();
    }
    float r = red[0];
    __syncthreads();
    return r;
}

__global__ void k_prep_rows(const float* __restrict__ pairs_text,
                            const float* __restrict__ att_l_w,
                            const float* __restrict__ att_l_b,
                            const float* __restrict__ att_r_w,
                            const float* __restrict__ att_r_b,
                            bf16* __restrict__ A_l, bf16* __restrict__ A_r,
                            float2* __restrict__ w01) {
    __shared__ float red[256];
    int m = blockIdx.x;
    const float* hl = pairs_text + (size_t)m * 2 * DD;
    const float* hr = hl + DD;
    float pl = 0.f, pr = 0.f;
    for (int d = threadIdx.x; d < DD; d += 256) {
        float xl = hl[d], xr = hr[d];
        A_l[(size_t)m * DD + d] = (bf16)xl;
        A_r[(size_t)m * DD + d] = (bf16)xr;
        pl += xl * att_l_w[d];
        pr += xr * att_r_w[d];
    }
    float wl = block_reduce256(pl, red);
    float wr = block_reduce256(pr, red);
    if (threadIdx.x == 0) {
        wl += att_l_b[0];
        wr += att_r_b[0];
        float mx = fmaxf(wl, wr);
        float e0 = __expf(wl - mx), e1 = __expf(wr - mx);
        float inv = 1.0f / (e0 + e1);
        w01[m] = make_float2(e0 * inv, e1 * inv);
    }
}

// ---------------------------------------------------------------------------
// Dual GEMM for composed:  C[m,n] = w0[m]*(Al@Wl^T + bl)[m,n]
//                                 + w1[m]*(Ar@Wr^T + br)[m,n]
// one wave per 16(M) x 32(N) tile;  grid = (M/16, N/32)
// ---------------------------------------------------------------------------
__global__ void __launch_bounds__(32)
k_gemm_composed(const bf16* __restrict__ Al, const bf16* __restrict__ Ar,
                const bf16* __restrict__ Wl, const bf16* __restrict__ Wr,
                const float* __restrict__ bl, const float* __restrict__ br,
                const float2* __restrict__ w01, float* __restrict__ C,
                int M, int N, int K) {
    int lane = threadIdx.x & 31;
    int l16 = lane & 15, hi = lane >> 4;
    int m0 = blockIdx.x * 16;
    int n0 = blockIdx.y * 32;

    f32x8 accl[2] = {}, accr[2] = {};
    for (int kb = 0; kb < K; kb += 32) {
        if (kb + 32 < K)
            __builtin_prefetch(Al + (size_t)(m0 + l16) * K + kb + 32, 0, 0);
        bf16x16 a_l = load_a_frag(Al, K, m0 + l16, kb, hi);
        bf16x16 a_r = load_a_frag(Ar, K, m0 + l16, kb, hi);
#pragma unroll
        for (int t = 0; t < 2; ++t) {
            bf16x16 b_l = load_b_frag(Wl, K, n0 + t * 16 + l16, kb, hi);
            bf16x16 b_r = load_b_frag(Wr, K, n0 + t * 16 + l16, kb, hi);
            accl[t] = __builtin_amdgcn_wmma_f32_16x16x32_bf16(
                false, a_l, false, b_l, (short)0, accl[t], false, false);
            accr[t] = __builtin_amdgcn_wmma_f32_16x16x32_bf16(
                false, a_r, false, b_r, (short)0, accr[t], false, false);
        }
    }
#pragma unroll
    for (int t = 0; t < 2; ++t) {
        int n = n0 + t * 16 + l16;
        float bln = bl[n], brn = br[n];
#pragma unroll
        for (int r = 0; r < 8; ++r) {
            int m = m0 + hi * 8 + r;
            float2 w = w01[m];
            C[(size_t)m * N + n] =
                w.x * (accl[t][r] + bln) + w.y * (accr[t][r] + brn);
        }
    }
}

// ---------------------------------------------------------------------------
// Generic WMMA GEMM:  C[m,n] = f(acc + bias[n]) (+ resid[m,n])
// one wave per 16(M) x 64(N) tile;  grid = (M/16, N/64)
// ACT: 0 none, 1 gelu ; OUTBF: write bf16 ; RESID/BIAS compile-time flags
// ---------------------------------------------------------------------------
template <int ACT, int OUTBF, int RESID, int BIAS>
__global__ void __launch_bounds__(32)
k_gemm(const bf16* __restrict__ A, const bf16* __restrict__ W,
       const float* __restrict__ bias, const float* __restrict__ resid,
       void* __restrict__ Cout, int M, int N, int K) {
    int lane = threadIdx.x & 31;
    int l16 = lane & 15, hi = lane >> 4;
    int m0 = blockIdx.x * 16;
    int n0 = blockIdx.y * 64;

    f32x8 acc[4] = {};
    for (int kb = 0; kb < K; kb += 32) {
        if (kb + 32 < K)
            __builtin_prefetch(A + (size_t)(m0 + l16) * K + kb + 32, 0, 0);
        bf16x16 a = load_a_frag(A, K, m0 + l16, kb, hi);
#pragma unroll
        for (int t = 0; t < 4; ++t) {
            bf16x16 b = load_b_frag(W, K, n0 + t * 16 + l16, kb, hi);
            acc[t] = __builtin_amdgcn_wmma_f32_16x16x32_bf16(
                false, a, false, b, (short)0, acc[t], false, false);
        }
    }
#pragma unroll
    for (int t = 0; t < 4; ++t) {
        int n = n0 + t * 16 + l16;
        float bn = BIAS ? bias[n] : 0.0f;
#pragma unroll
        for (int r = 0; r < 8; ++r) {
            int m = m0 + hi * 8 + r;
            float v = acc[t][r] + bn;
            if (ACT == 1) v = geluf(v);
            if (RESID) v += resid[(size_t)m * N + n];
            if (OUTBF)
                ((bf16*)Cout)[(size_t)m * N + n] = (bf16)v;
            else
                ((float*)Cout)[(size_t)m * N + n] = v;
        }
    }
}

// ---------------------------------------------------------------------------
// LayerNorm per row of 768;  one block per row
// ---------------------------------------------------------------------------
__global__ void k_layernorm(const float* __restrict__ x,
                            const float* __restrict__ g,
                            const float* __restrict__ b,
                            float* __restrict__ h) {
    __shared__ float red[256];
    int m = blockIdx.x;
    const float* xr = x + (size_t)m * DD;
    float s = 0.f, s2 = 0.f;
    for (int d = threadIdx.x; d < DD; d += 256) {
        float v = xr[d];
        s += v;
        s2 += v * v;
    }
    float sum = block_reduce256(s, red);
    float sumsq = block_reduce256(s2, red);
    float mu = sum * (1.0f / DD);
    float var = sumsq * (1.0f / DD) - mu * mu;
    float inv = rsqrtf(var + 1e-12f);
    for (int d = threadIdx.x; d < DD; d += 256)
        h[(size_t)m * DD + d] = (xr[d] - mu) * inv * g[d] + b[d];
}

// ---------------------------------------------------------------------------
// energies -> softmax over NS=32 -> span ;  one block per (b,s)
// ---------------------------------------------------------------------------
__global__ void k_span(const float* __restrict__ h,
                       const float* __restrict__ energy_w,
                       const float* __restrict__ energy_b,
                       float* __restrict__ span) {
    __shared__ float red[256];
    __shared__ float e[32];
    int bs = blockIdx.x;
    const float* hb = h + (size_t)bs * 32 * DD;
    for (int ns = 0; ns < 32; ++ns) {
        float p = 0.f;
        for (int d = threadIdx.x; d < DD; d += 256)
            p += hb[(size_t)ns * DD + d] * energy_w[d];
        float dot = block_reduce256(p, red);
        if (threadIdx.x == 0) e[ns] = dot + energy_b[0];
    }
    __syncthreads();
    if (threadIdx.x == 0) {
        float mx = -1e30f;
        for (int ns = 0; ns < 32; ++ns) mx = fmaxf(mx, e[ns]);
        float den = 0.f;
        for (int ns = 0; ns < 32; ++ns) {
            e[ns] = __expf(e[ns] - mx);
            den += e[ns];
        }
        float inv = 1.0f / den;
        for (int ns = 0; ns < 32; ++ns) e[ns] *= inv;
    }
    __syncthreads();
    for (int d = threadIdx.x; d < DD; d += 256) {
        float acc = 0.f;
        for (int ns = 0; ns < 32; ++ns) acc += e[ns] * hb[(size_t)ns * DD + d];
        span[(size_t)bs * DD + d] = acc;
    }
}

// ---------------------------------------------------------------------------
// left-branch input (B,S,O,D) packed bf16
// ---------------------------------------------------------------------------
__global__ void k_prep_left(const float* __restrict__ vis_emb,
                            const float* __restrict__ span,
                            const float* __restrict__ pairs_den,
                            const float* __restrict__ same_obj_emb,
                            const float* __restrict__ den_emb,
                            bf16* __restrict__ A_left) {
    long i = (long)blockIdx.x * blockDim.x + threadIdx.x;  // (bs*64+o)*768+d
    long row = i / DD;
    int d = (int)(i % DD);
    int o = (int)(row & 63);
    long bs = row >> 6;
    int b = (int)(bs >> 7);
    float ld = pairs_den[(bs * 2 + 0) * 64 + o];
    float rd = pairs_den[(bs * 2 + 1) * 64 + o];
    float v = vis_emb[((size_t)b * 64 + o) * DD + d] + span[bs * DD + d] +
              same_obj_emb[d] * rd + same_obj_emb[DD + d] * (1.0f - rd) +
              den_emb[d] * ld + den_emb[DD + d] * (1.0f - ld);
    A_left[i] = (bf16)v;
}

// right-branch input (B,S,D): right_den @ vis_emb + span, packed bf16
__global__ void k_prep_right(const float* __restrict__ vis_emb,
                             const float* __restrict__ span,
                             const float* __restrict__ pairs_den,
                             bf16* __restrict__ A_right) {
    long i = (long)blockIdx.x * blockDim.x + threadIdx.x;  // bs*768+d
    long bs = i / DD;
    int d = (int)(i % DD);
    int b = (int)(bs >> 7);
    float acc = 0.f;
    for (int o = 0; o < 64; ++o)
        acc += pairs_den[(bs * 2 + 1) * 64 + o] *
               vis_emb[((size_t)b * 64 + o) * DD + d];
    A_right[i] = (bf16)(acc + span[i]);
}

// c[bs] = ff3_b . right_m[bs,:]
__global__ void k_cvec(const float* __restrict__ b3,
                       const float* __restrict__ right_m,
                       float* __restrict__ c) {
    __shared__ float red[256];
    int bs = blockIdx.x;
    float p = 0.f;
    for (int d = threadIdx.x; d < DD; d += 256)
        p += b3[d] * right_m[(size_t)bs * DD + d];
    float s = block_reduce256(p, red);
    if (threadIdx.x == 0) c[bs] = s;
}

// den[b,s,o] = sigmoid(g1[bso,:] . t[bs,:] + c[bs]) * obj_mask[b,o]
__global__ void k_final(const bf16* __restrict__ g1,
                        const float* __restrict__ t,
                        const float* __restrict__ c,
                        const float* __restrict__ obj_mask,
                        float* __restrict__ out) {
    __shared__ float red[256];
    long row = blockIdx.x;  // bs*64+o
    long bs = row >> 6;
    int o = (int)(row & 63);
    int b = (int)(bs >> 7);
    float p = 0.f;
    for (int d = threadIdx.x; d < DD; d += 256)
        p += (float)g1[row * DD + d] * t[bs * DD + d];
    float dot = block_reduce256(p, red);
    if (threadIdx.x == 0)
        out[row] = sigmoidf_(dot + c[bs]) * obj_mask[(size_t)b * 64 + o];
}

// ---------------------------------------------------------------------------
// Launch
// ---------------------------------------------------------------------------
extern "C" void kernel_launch(void* const* d_in, const int* in_sizes, int n_in,
                              void* d_out, int out_size, void* d_ws,
                              size_t ws_size, hipStream_t stream) {
    const float* pairs_text = (const float*)d_in[0];
    const float* vis_emb    = (const float*)d_in[1];
    const float* pairs_den  = (const float*)d_in[2];
    const float* obj_mask   = (const float*)d_in[3];
    const float* att_l_w = (const float*)d_in[4];
    const float* att_l_b = (const float*)d_in[5];
    const float* att_r_w = (const float*)d_in[6];
    const float* att_r_b = (const float*)d_in[7];
    const float* lin_l_w = (const float*)d_in[8];
    const float* lin_l_b = (const float*)d_in[9];
    const float* lin_r_w = (const float*)d_in[10];
    const float* lin_r_b = (const float*)d_in[11];
    const float* inter_w = (const float*)d_in[12];
    const float* inter_b = (const float*)d_in[13];
    const float* ln_g    = (const float*)d_in[14];
    const float* ln_b    = (const float*)d_in[15];
    const float* energy_w = (const float*)d_in[16];
    const float* energy_b = (const float*)d_in[17];
    const float* ff1_w = (const float*)d_in[18];
    const float* ff1_b = (const float*)d_in[19];
    const float* ff2_w = (const float*)d_in[20];
    const float* ff2_b = (const float*)d_in[21];
    const float* ff3_w = (const float*)d_in[22];
    const float* ff3_b = (const float*)d_in[23];
    const float* ff4_w = (const float*)d_in[24];
    const float* ff4_b = (const float*)d_in[25];
    const float* same_obj_emb = (const float*)d_in[26];
    const float* den_emb      = (const float*)d_in[27];

    const int Mrows = 32768;   // B*S*NS
    const int Mleft = 65536;   // B*S*O
    const int BS    = 1024;    // B*S

    // ---- workspace layout (bytes) ----
    char* ws = (char*)d_ws;
    bf16* WBl  = (bf16*)(ws + 0 * (size_t)DD2 * 2);
    bf16* WBr  = (bf16*)(ws + 1 * (size_t)DD2 * 2);
    bf16* WBi  = (bf16*)(ws + 2 * (size_t)DD2 * 2);
    bf16* WB1  = (bf16*)(ws + 3 * (size_t)DD2 * 2);
    bf16* WB2  = (bf16*)(ws + 4 * (size_t)DD2 * 2);
    bf16* WB4  = (bf16*)(ws + 5 * (size_t)DD2 * 2);
    bf16* WB3T = (bf16*)(ws + 6 * (size_t)DD2 * 2);
    size_t off = 7 * (size_t)DD2 * 2;                       // 8,257,536
    float2* w01 = (float2*)(ws + off);  off += (size_t)Mrows * 8;
    bf16* A_l = (bf16*)(ws + off);      // also reused as cbf16
    bf16* A_r = (bf16*)(ws + off + (size_t)Mrows * DD * 2);
    off += 2 * (size_t)Mrows * DD * 2;
    float* composed = (float*)(ws + off);  // later: h, then g1 (bf16)
    off += (size_t)Mrows * DD * 4;
    float* xbuf = (float*)(ws + off);      // later: A_left (bf16, same bytes)
    off += (size_t)Mrows * DD * 4;
    float* span = (float*)(ws + off);      off += (size_t)BS * DD * 4;
    bf16* A_right = (bf16*)(ws + off);     off += (size_t)BS * DD * 2;
    bf16* g2 = (bf16*)(ws + off);          off += (size_t)BS * DD * 2;
    float* right_m = (float*)(ws + off);   off += (size_t)BS * DD * 4;
    bf16* A_rm = (bf16*)(ws + off);        off += (size_t)BS * DD * 2;
    float* tbuf = (float*)(ws + off);      off += (size_t)BS * DD * 4;
    float* cvec = (float*)(ws + off);      off += 4096;
    bf16* cbf16 = A_l;                 // composed in bf16 (A_l dead)
    float* hbuf = composed;            // LN output overwrites composed
    bf16* g1 = (bf16*)composed;        // gelu(ff1) output (h dead)
    bf16* A_left = (bf16*)xbuf;        // left-branch rows (xbuf dead)

    // 1. pack weights to bf16 (+ transposed ff3)
    {
        int blk = (DD2 + 255) / 256;
        k_f32_to_bf16<<<blk, 256, 0, stream>>>(lin_l_w, WBl, DD2);
        k_f32_to_bf16<<<blk, 256, 0, stream>>>(lin_r_w, WBr, DD2);
        k_f32_to_bf16<<<blk, 256, 0, stream>>>(inter_w, WBi, DD2);
        k_f32_to_bf16<<<blk, 256, 0, stream>>>(ff1_w, WB1, DD2);
        k_f32_to_bf16<<<blk, 256, 0, stream>>>(ff2_w, WB2, DD2);
        k_f32_to_bf16<<<blk, 256, 0, stream>>>(ff4_w, WB4, DD2);
        k_transpose_bf16<<<blk, 256, 0, stream>>>(ff3_w, WB3T);
    }

    // 2. gating weights + pack hs_l/hs_r
    k_prep_rows<<<Mrows, 256, 0, stream>>>(pairs_text, att_l_w, att_l_b,
                                           att_r_w, att_r_b, A_l, A_r, w01);

    // 3. composed = w0*(Al@Wl^T+bl) + w1*(Ar@Wr^T+br)
    k_gemm_composed<<<dim3(Mrows / 16, DD / 32), 32, 0, stream>>>(
        A_l, A_r, WBl, WBr, lin_l_b, lin_r_b, w01, composed, Mrows, DD, DD);

    // 4. composed -> bf16
    k_f32_to_bf16<<<(int)(((long)Mrows * DD + 255) / 256), 256, 0, stream>>>(
        composed, cbf16, (long)Mrows * DD);

    // 5. x = gelu(composed@Wi^T + bi) + composed
    k_gemm<1, 0, 1, 1><<<dim3(Mrows / 16, DD / 64), 32, 0, stream>>>(
        cbf16, WBi, inter_b, composed, xbuf, Mrows, DD, DD);

    // 6. h = LN(x)
    k_layernorm<<<Mrows, 256, 0, stream>>>(xbuf, ln_g, ln_b, hbuf);

    // 7. span attention
    k_span<<<BS, 256, 0, stream>>>(hbuf, energy_w, energy_b, span);

    // 8/9. branch inputs
    k_prep_left<<<(int)(((long)Mleft * DD) / 256), 256, 0, stream>>>(
        vis_emb, span, pairs_den, same_obj_emb, den_emb, A_left);
    k_prep_right<<<(int)(((long)BS * DD) / 256), 256, 0, stream>>>(
        vis_emb, span, pairs_den, A_right);

    // 10. g1 = gelu(A_left@ff1^T + b1)   (bf16 out)
    k_gemm<1, 1, 0, 1><<<dim3(Mleft / 16, DD / 64), 32, 0, stream>>>(
        A_left, WB1, ff1_b, nullptr, g1, Mleft, DD, DD);

    // 11. g2 = gelu(A_right@ff2^T + b2)  (bf16 out)
    k_gemm<1, 1, 0, 1><<<dim3(BS / 16, DD / 64), 32, 0, stream>>>(
        A_right, WB2, ff2_b, nullptr, g2, BS, DD, DD);

    // 12. right_m = g2@ff4^T + b4  (f32)
    k_gemm<0, 0, 0, 1><<<dim3(BS / 16, DD / 64), 32, 0, stream>>>(
        g2, WB4, ff4_b, nullptr, right_m, BS, DD, DD);

    // 13/14. t = right_m @ ff3  (factored ff3 GEMM over 1024 rows only)
    k_f32_to_bf16<<<(int)(((long)BS * DD + 255) / 256), 256, 0, stream>>>(
        right_m, A_rm, (long)BS * DD);
    k_gemm<0, 0, 0, 0><<<dim3(BS / 16, DD / 64), 32, 0, stream>>>(
        A_rm, WB3T, nullptr, nullptr, tbuf, BS, DD, DD);

    // 15. c[bs] = b3 . right_m
    k_cvec<<<BS, 256, 0, stream>>>(ff3_b, right_m, cvec);

    // 16. den = sigmoid(g1 . t + c) * mask
    k_final<<<Mleft, 256, 0, stream>>>(g1, tbuf, cvec, obj_mask,
                                       (float*)d_out);
}
